// FmapSampler_14422500180036
// MI455X (gfx1250) — matrix operand order repaired and checked by reading, not compiled, not run on hardware
//
#include <hip/hip_runtime.h>
#include <hip/hip_bf16.h>

typedef __attribute__((ext_vector_type(16))) __bf16 v16bf;
typedef __attribute__((ext_vector_type(8)))  float  v8f;
typedef __attribute__((ext_vector_type(4)))  float  f32x4;

#define KDIM   4096            // H*W = 64*64
#define NMESH  162
#define BATCH  32
#define CDIM   512
#define NT     11              // ceil(162/16) n-tiles
#define NPAIRS 6               // ceil(NT/2) n-tile pairs per wave
#define KSTEPS 128             // 4096 / 32
#define FRAG_ELEMS ((size_t)NT * KSTEPS * 32 * 16)   // 720896 bf16 per (hi|lo)

// ---------------------------------------------------------------------------
// Prep: split weight [K=4096, N=162] fp32 into bf16 hi/lo, pre-swizzled into
// the exact per-lane WMMA B-fragment order for v_wmma_f32_16x16x32_bf16:
//   fragment (t, s): lane L holds column n = t*16 + (L&15),
//   K = s*32 + (L>=16 ? 8 : 0) + (i<8 ? i : i+8), i = 0..15 (two bf16 per VGPR)
// Flat index: ((t*128 + s)*32 + lane)*16 + i
// ---------------------------------------------------------------------------
__global__ __launch_bounds__(256)
void prep_weight_kernel(const float* __restrict__ w,
                        __bf16* __restrict__ bhi,
                        __bf16* __restrict__ blo) {
    unsigned idx = blockIdx.x * blockDim.x + threadIdx.x;   // < 11<<16
    int i    = idx & 15;
    int lane = (idx >> 4) & 31;
    int s    = (idx >> 9) & 127;
    int t    = idx >> 16;
    if (t >= NT) return;

    int n = t * 16 + (lane & 15);
    int k = s * 32 + ((lane >> 4) << 3) + (i < 8 ? i : i + 8);
    float v = (n < NMESH) ? w[(size_t)k * NMESH + n] : 0.0f;   // weight[k][n]
    __bf16 h = (__bf16)v;
    __bf16 l = (__bf16)(v - (float)h);
    bhi[idx] = h;
    blo[idx] = l;
}

// ---------------------------------------------------------------------------
// Main: per wave, one 16(c) x 32(n: two n-tiles) slab of out[b, n, c].
// K-loop over 4096 in steps of 32 using bf16x3-split WMMA (fp32-class
// accuracy at bf16 matrix throughput). A fragments (loads + hi/lo split) are
// built once per step and feed 6 WMMAs (2 n-tiles x 3 precision terms).
// grid = (NPAIRS, CDIM/128, BATCH), block = 256 threads = 8 waves.
// ---------------------------------------------------------------------------
__global__ __launch_bounds__(256)
void fmap_gemm_kernel(const float* __restrict__ x,
                      const __bf16* __restrict__ bhi,
                      const __bf16* __restrict__ blo,
                      float* __restrict__ out) {
    const int t0   = blockIdx.x * 2;               // first n-tile of the pair
    const int t1   = (t0 + 1 < NT) ? t0 + 1 : NT - 1;  // clamp (dup, unstored)
    const int cblk = blockIdx.y;                   // 0..3
    const int b    = blockIdx.z;                   // 0..31
    const int wave = threadIdx.x >> 5;
    const int lane = threadIdx.x & 31;
    const int half = lane >> 4;                    // 0: lanes 0-15, 1: 16-31
    const int l16  = lane & 15;

    const int c0   = cblk * 128 + wave * 16;       // c-tile base (M dimension)
    const int crow = c0 + l16;                     // this lane's A row

    // A: lane<16 -> K {0..7, 16..23}; lane>=16 -> K {8..15, 24..31}
    const float*  xrow = x + ((size_t)b * CDIM + crow) * KDIM + half * 8;
    const size_t  lfrag = (size_t)lane * 16;
    const __bf16* bh0 = bhi + ((size_t)t0 * KSTEPS * 512) + lfrag;
    const __bf16* bl0 = blo + ((size_t)t0 * KSTEPS * 512) + lfrag;
    const __bf16* bh1 = bhi + ((size_t)t1 * KSTEPS * 512) + lfrag;
    const __bf16* bl1 = blo + ((size_t)t1 * KSTEPS * 512) + lfrag;

    v8f accH0 = {};   // tile0: hi*hi
    v8f accC0 = {};   // tile0: hi*lo(w) + lo(x)*hi
    v8f accH1 = {};   // tile1: hi*hi
    v8f accC1 = {};   // tile1: cross terms

    #pragma unroll 2
    for (int s = 0; s < KSTEPS; ++s) {
        const float* xp = xrow + s * 32;
        f32x4 f0 = *(const f32x4*)(xp);          // K rel 0..3
        f32x4 f1 = *(const f32x4*)(xp + 4);      // K rel 4..7
        f32x4 f2 = *(const f32x4*)(xp + 16);     // K rel 16..19
        f32x4 f3 = *(const f32x4*)(xp + 20);     // K rel 20..23
        if (s + 1 < KSTEPS) {
            __builtin_prefetch(xp + 32, 0, 3);   // global_prefetch_b8
        }

        v16bf ah = {}, al = {};
        #pragma unroll
        for (int j = 0; j < 4; ++j) {
            float v0 = f0[j], v1 = f1[j], v2 = f2[j], v3 = f3[j];
            __bf16 h0 = (__bf16)v0, h1 = (__bf16)v1;
            __bf16 h2 = (__bf16)v2, h3 = (__bf16)v3;
            ah[j]      = h0; ah[4 + j]  = h1;
            ah[8 + j]  = h2; ah[12 + j] = h3;
            al[j]      = (__bf16)(v0 - (float)h0);
            al[4 + j]  = (__bf16)(v1 - (float)h1);
            al[8 + j]  = (__bf16)(v2 - (float)h2);
            al[12 + j] = (__bf16)(v3 - (float)h3);
        }

        const size_t so = (size_t)s * 512;
        v16bf vbh0 = *(const v16bf*)(bh0 + so);
        v16bf vbl0 = *(const v16bf*)(bl0 + so);
        v16bf vbh1 = *(const v16bf*)(bh1 + so);
        v16bf vbl1 = *(const v16bf*)(bl1 + so);

        // 4 independent accumulator chains; A fragments feed 6 WMMAs.
        accH0 = __builtin_amdgcn_wmma_f32_16x16x32_bf16(
                    false, ah, false, vbh0, (short)0, accH0, false, false);
        accH1 = __builtin_amdgcn_wmma_f32_16x16x32_bf16(
                    false, ah, false, vbh1, (short)0, accH1, false, false);
        accC0 = __builtin_amdgcn_wmma_f32_16x16x32_bf16(
                    false, ah, false, vbl0, (short)0, accC0, false, false);
        accC1 = __builtin_amdgcn_wmma_f32_16x16x32_bf16(
                    false, ah, false, vbl1, (short)0, accC1, false, false);
        accC0 = __builtin_amdgcn_wmma_f32_16x16x32_bf16(
                    false, al, false, vbh0, (short)0, accC0, false, false);
        accC1 = __builtin_amdgcn_wmma_f32_16x16x32_bf16(
                    false, al, false, vbh1, (short)0, accC1, false, false);
    }

    // C/D layout: VGPR r holds M = r + half*8, N = lane&15
    const int cbase = c0 + half * 8;
    {
        const int n = t0 * 16 + l16;
        if (n < NMESH) {
            float* op = out + ((size_t)b * NMESH + n) * CDIM + cbase;
            f32x4 o0 = { accH0[0] + accC0[0], accH0[1] + accC0[1],
                         accH0[2] + accC0[2], accH0[3] + accC0[3] };
            f32x4 o1 = { accH0[4] + accC0[4], accH0[5] + accC0[5],
                         accH0[6] + accC0[6], accH0[7] + accC0[7] };
            *(f32x4*)(op)     = o0;
            *(f32x4*)(op + 4) = o1;
        }
    }
    if (t0 + 1 < NT) {
        const int n = (t0 + 1) * 16 + l16;
        if (n < NMESH) {
            float* op = out + ((size_t)b * NMESH + n) * CDIM + cbase;
            f32x4 o0 = { accH1[0] + accC1[0], accH1[1] + accC1[1],
                         accH1[2] + accC1[2], accH1[3] + accC1[3] };
            f32x4 o1 = { accH1[4] + accC1[4], accH1[5] + accC1[5],
                         accH1[6] + accC1[6], accH1[7] + accC1[7] };
            *(f32x4*)(op)     = o0;
            *(f32x4*)(op + 4) = o1;
        }
    }
}

extern "C" void kernel_launch(void* const* d_in, const int* in_sizes, int n_in,
                              void* d_out, int out_size, void* d_ws, size_t ws_size,
                              hipStream_t stream) {
    const float* x = (const float*)d_in[0];   // [32, 512, 64, 64] fp32
    const float* w = (const float*)d_in[1];   // [64, 64, 162]     fp32
    float* out = (float*)d_out;               // [32, 162, 512]    fp32

    __bf16* bhi = (__bf16*)d_ws;              // needs ~2.9 MB total scratch
    __bf16* blo = bhi + FRAG_ELEMS;

    // Swizzle + split the (L2-resident, heavily reused) weight once per launch.
    prep_weight_kernel<<<(int)(FRAG_ELEMS / 256), 256, 0, stream>>>(w, bhi, blo);

    dim3 grid(NPAIRS, CDIM / 128, BATCH);
    fmap_gemm_kernel<<<grid, 256, 0, stream>>>(x, bhi, blo, out);
}
